// DNRI_Decoder_49357764166097
// MI455X (gfx1250) — compile-verified, matrix operand order checked by reading, not run on hardware
//
#include <hip/hip_runtime.h>
#include <hip/hip_bf16.h>
#include <math.h>

#define BATCH   64
#define NVARS   50
#define H       256
#define D       64
#define TWOH    512
#define NEDGE   2450                  // 50*49
#define ROWS_E  (BATCH * NEDGE)       // 156800 edge rows
#define ROWS_N  (BATCH * NVARS)       // 3200 node rows

typedef __attribute__((ext_vector_type(16))) __bf16 v16bf;
typedef __attribute__((ext_vector_type(8)))  __bf16 v8bf;
typedef __attribute__((ext_vector_type(8)))  float  v8f;

// ---------------------------------------------------------------------------
// Fast activations: gfx1250 has a hardware V_TANH_F32 (TRANS group, which
// co-executes with XDL/WMMA ops). sigmoid(x) == 0.5*tanh(x/2)+0.5 exactly.
// ---------------------------------------------------------------------------
__device__ __forceinline__ float fast_tanh(float x) {
#if __has_builtin(__builtin_amdgcn_tanhf)
  return __builtin_amdgcn_tanhf(x);
#else
  return tanhf(x);
#endif
}
__device__ __forceinline__ float fast_sig(float x) {
  return fmaf(0.5f, fast_tanh(0.5f * x), 0.5f);
}

// ---------------------------------------------------------------------------
// Fragment load for v_wmma_f32_16x16x32_bf16.
// A-fragment: rowp = base of this lane's matrix row (row = lane&15), holds
//   K = {k0+off .. k0+off+7, k0+16+off .. k0+16+off+7}, off = 8*(lane>=16).
// B-fragment (column-major usage): rowp = base of this lane's output column
//   (col = lane&15) in a row-major [N][K] weight matrix; same K pattern.
// Two 16-byte loads -> global_load_b128 / ds_load_b128.
// ---------------------------------------------------------------------------
__device__ __forceinline__ v16bf ld_frag(const __bf16* rowp, int k0) {
  const int hi8 = ((threadIdx.x & 31) >> 4) << 3;   // 0 or 8
  const __bf16* p = rowp + k0 + hi8;
  v8bf lo = *(const v8bf*)(p);
  v8bf hh = *(const v8bf*)(p + 16);
  v16bf f;
#pragma unroll
  for (int i = 0; i < 8; ++i) { f[i] = lo[i]; f[i + 8] = hh[i]; }
  return f;
}

__device__ __forceinline__ v8f wmma_bf(v16bf a, v16bf b, v8f c) {
  // (neg_a, A, neg_b, B, c_mod, C, reuse_a, reuse_b)
  return __builtin_amdgcn_wmma_f32_16x16x32_bf16(false, a, false, b,
                                                 (short)0, c, false, false);
}

// C fragment element j of a v8f maps to (row = j + 8*(lane>=16), col = lane&15).

__global__ void k_f32_to_bf16(const float* __restrict__ s,
                              __bf16* __restrict__ d, int n) {
  int i = blockIdx.x * blockDim.x + threadIdx.x;
  if (i < n) d[i] = (__bf16)s[i];
}

__global__ void k_zero(float* __restrict__ p, int n) {
  int i = blockIdx.x * blockDim.x + threadIdx.x;
  if (i < n) p[i] = 0.f;
}

// ---------------------------------------------------------------------------
// Edge-message MLP: for k=1..3
//   m = tanh(tanh(pre_msg @ W1[k]^T + b1[k]) @ W2[k]^T + b2[k]) * edges[:,:,k]/3
// summed over k, then scatter-added (with /49) into agg[b, recv, :].
// One wave owns 16 consecutive edge rows. pre_msg is formed implicitly:
// K<256 -> hidden[b,recv], K>=256 -> hidden[b,send].
// ---------------------------------------------------------------------------
__global__ __launch_bounds__(64) void k_edge_msgs(
    const __bf16* __restrict__ hid,
    const __bf16* __restrict__ w1, const float* __restrict__ b1,
    const __bf16* __restrict__ w2, const float* __restrict__ b2,
    const float*  __restrict__ edges,
    float* __restrict__ agg)
{
  __shared__ __attribute__((aligned(16))) __bf16 m1s[2][16 * H];  // 8 KB/wave
  __shared__ __attribute__((aligned(16))) float  msgs[2][16 * H]; // 16 KB/wave
  const int lane = threadIdx.x & 31;
  const int wid  = threadIdx.x >> 5;
  const int tile = blockIdx.x * 2 + wid;
  const int row0 = tile * 16;
  const int hi   = lane >> 4;
  const int lm   = lane & 15;
  const v8f vzero = {0.f, 0.f, 0.f, 0.f, 0.f, 0.f, 0.f, 0.f};

  // per-lane A row pointers (row = row0 + lm)
  const int rA = row0 + lm;
  const int bA = rA / NEDGE;
  const int eA = rA - bA * NEDGE;
  const int sA = eA / (NVARS - 1);
  const int jA = eA - sA * (NVARS - 1);
  const int vA = jA + (jA >= sA ? 1 : 0);
  const __bf16* pR = hid + (size_t)(bA * NVARS + vA) * H;
  const __bf16* pS = hid + (size_t)(bA * NVARS + sA) * H;

#pragma unroll 1
  for (int k = 1; k < 4; ++k) {
    const __bf16* W1k = w1 + (size_t)k * H * TWOH;
    const __bf16* W2k = w2 + (size_t)k * H * H;
    const float*  b1k = b1 + k * H;
    const float*  b2k = b2 + k * H;
    __builtin_prefetch(W2k, 0, 0);   // pull fc2 weights toward cache

    v8f acc[16];
#pragma unroll
    for (int n = 0; n < 16; ++n) acc[n] = vzero;

    // ---- fc1: [16 x 512] x [512 -> 256] ----
#pragma unroll 1
    for (int kt = 0; kt < 16; ++kt) {
      const __bf16* ap = (kt < 8) ? pR : pS;
      v16bf a = ld_frag(ap, (kt & 7) * 32);
#pragma unroll
      for (int n = 0; n < 16; ++n) {
        v16bf bb = ld_frag(W1k + (size_t)(n * 16 + lm) * TWOH, kt * 32);
        acc[n] = wmma_bf(a, bb, acc[n]);
      }
    }
    // bias + tanh -> LDS bf16 (A matrix for fc2)
#pragma unroll
    for (int n = 0; n < 16; ++n) {
#pragma unroll
      for (int j = 0; j < 8; ++j) {
        const int rr = j + hi * 8;
        const int cc = n * 16 + lm;
        m1s[wid][rr * H + cc] = (__bf16)fast_tanh(acc[n][j] + b1k[cc]);
        acc[n][j] = 0.f;
      }
    }
    __syncthreads();

    // ---- fc2: [16 x 256] x [256 -> 256] ----
#pragma unroll 1
    for (int kt = 0; kt < 8; ++kt) {
      v16bf a = ld_frag(&m1s[wid][lm * H], kt * 32);
#pragma unroll
      for (int n = 0; n < 16; ++n) {
        v16bf bb = ld_frag(W2k + (size_t)(n * 16 + lm) * H, kt * 32);
        acc[n] = wmma_bf(a, bb, acc[n]);
      }
    }
    // tanh, edge gate (/3 for SKIP_FIRST norm), accumulate in wave-private LDS
    float ef[8];
#pragma unroll
    for (int j = 0; j < 8; ++j)
      ef[j] = edges[(size_t)(row0 + j + hi * 8) * 4 + k] * (1.f / 3.f);
#pragma unroll
    for (int n = 0; n < 16; ++n) {
#pragma unroll
      for (int j = 0; j < 8; ++j) {
        const int rr = j + hi * 8;
        const int cc = n * 16 + lm;
        const float v = fast_tanh(acc[n][j] + b2k[cc]) * ef[j];
        if (k == 1) msgs[wid][rr * H + cc] = v;
        else        msgs[wid][rr * H + cc] += v;
      }
    }
    __syncthreads();
  }

  // ---- scatter-add to receiver nodes, with /(V-1) ----
#pragma unroll 1
  for (int rr = 0; rr < 16; ++rr) {
    const int rg = row0 + rr;
    const int b  = rg / NEDGE;
    const int e  = rg - b * NEDGE;
    const int s  = e / (NVARS - 1);
    const int j  = e - s * (NVARS - 1);
    const int rv = j + (j >= s ? 1 : 0);
    float* dst = agg + (size_t)(b * NVARS + rv) * H;
    for (int c = lane; c < H; c += 32)
      atomicAdd(&dst[c], msgs[wid][rr * H + c] * (1.f / 49.f));
  }
}

// ---------------------------------------------------------------------------
// GRU-style node update. One wave owns a 16-row tile of [B*V, H].
// r = sig(inp@ir^T + irb + agg@hr^T);  i = sig(inp@ii^T + iib + agg@hi^T)
// n = tanh(inp@in^T + inb + r * (agg@hh^T));  nh = (1-i)*n + i*hidden
// ---------------------------------------------------------------------------
__global__ __launch_bounds__(64) void k_gru(
    const float*  __restrict__ agg,  const __bf16* __restrict__ inpbf,
    const float*  __restrict__ hidden,
    const __bf16* __restrict__ hr,   const __bf16* __restrict__ hiw,
    const __bf16* __restrict__ hh,
    const __bf16* __restrict__ ir,   const float* __restrict__ irb,
    const __bf16* __restrict__ ii,   const float* __restrict__ iib,
    const __bf16* __restrict__ inw,  const float* __restrict__ inb,
    float* __restrict__ nh_out, __bf16* __restrict__ nhbf)
{
  __shared__ __attribute__((aligned(16))) __bf16 aggs[2][16 * H];
  const int lane = threadIdx.x & 31;
  const int wid  = threadIdx.x >> 5;
  const int tile = blockIdx.x * 2 + wid;
  const int row0 = tile * 16;
  const int hi = lane >> 4, lm = lane & 15;
  const v8f vzero = {0.f, 0.f, 0.f, 0.f, 0.f, 0.f, 0.f, 0.f};

  for (int i = lane; i < 16 * H; i += 32)
    aggs[wid][i] = (__bf16)agg[(size_t)row0 * H + i];
  __syncthreads();

  const __bf16* arow = &aggs[wid][lm * H];
  const __bf16* xrow = inpbf + (size_t)(row0 + lm) * D;

#pragma unroll 1
  for (int nc = 0; nc < 4; ++nc) {                // 4 chunks of 64 cols
    v8f accr[4], acci[4], acch[4], accn[4];
#pragma unroll
    for (int n = 0; n < 4; ++n) { accr[n]=vzero; acci[n]=vzero; acch[n]=vzero; accn[n]=vzero; }

    // input-side GEMMs (K = 64)
#pragma unroll 1
    for (int kt = 0; kt < 2; ++kt) {
      v16bf a = ld_frag(xrow, kt * 32);
#pragma unroll
      for (int n = 0; n < 4; ++n) {
        const int col = nc * 64 + n * 16 + lm;
        accr[n] = wmma_bf(a, ld_frag(ir  + (size_t)col * D, kt * 32), accr[n]);
        acci[n] = wmma_bf(a, ld_frag(ii  + (size_t)col * D, kt * 32), acci[n]);
        accn[n] = wmma_bf(a, ld_frag(inw + (size_t)col * D, kt * 32), accn[n]);
      }
    }
    // agg-side GEMMs (K = 256)
#pragma unroll 1
    for (int kt = 0; kt < 8; ++kt) {
      v16bf a = ld_frag(arow, kt * 32);
#pragma unroll
      for (int n = 0; n < 4; ++n) {
        const int col = nc * 64 + n * 16 + lm;
        accr[n] = wmma_bf(a, ld_frag(hr  + (size_t)col * H, kt * 32), accr[n]);
        acci[n] = wmma_bf(a, ld_frag(hiw + (size_t)col * H, kt * 32), acci[n]);
        acch[n] = wmma_bf(a, ld_frag(hh  + (size_t)col * H, kt * 32), acch[n]);
      }
    }
    // fused gates
#pragma unroll
    for (int n = 0; n < 4; ++n) {
#pragma unroll
      for (int j = 0; j < 8; ++j) {
        const int rr  = j + hi * 8;
        const int col = nc * 64 + n * 16 + lm;
        const size_t gi = (size_t)(row0 + rr) * H + col;
        const float rv = fast_sig(accr[n][j] + irb[col]);
        const float iv = fast_sig(acci[n][j] + iib[col]);
        const float nv = fast_tanh(accn[n][j] + inb[col] + rv * acch[n][j]);
        const float nh = (1.f - iv) * nv + iv * hidden[gi];
        nh_out[gi] = nh;
        nhbf[gi]   = (__bf16)nh;
      }
    }
  }
}

// ---------------------------------------------------------------------------
// Output head: pred1 = relu(nh@of1^T+b); pred = relu(pred1@of2^T+b);
// mu = pred@mu^T + b.  One wave per 16-row tile.
// ---------------------------------------------------------------------------
__global__ __launch_bounds__(64) void k_outhead(
    const __bf16* __restrict__ nhbf,
    const __bf16* __restrict__ of1, const float* __restrict__ of1b,
    const __bf16* __restrict__ of2, const float* __restrict__ of2b,
    const __bf16* __restrict__ muw, const float* __restrict__ mub,
    float* __restrict__ pred_out, float* __restrict__ mu_out)
{
  __shared__ __attribute__((aligned(16))) __bf16 p1s[2][16 * H];
  __shared__ __attribute__((aligned(16))) __bf16 p2s[2][16 * H];
  const int lane = threadIdx.x & 31;
  const int wid  = threadIdx.x >> 5;
  const int tile = blockIdx.x * 2 + wid;
  const int row0 = tile * 16;
  const int hi = lane >> 4, lm = lane & 15;
  const v8f vzero = {0.f, 0.f, 0.f, 0.f, 0.f, 0.f, 0.f, 0.f};

  const __bf16* nrow = nhbf + (size_t)(row0 + lm) * H;
  v8f acc[16];
#pragma unroll
  for (int n = 0; n < 16; ++n) acc[n] = vzero;

  // of1
#pragma unroll 1
  for (int kt = 0; kt < 8; ++kt) {
    v16bf a = ld_frag(nrow, kt * 32);
#pragma unroll
    for (int n = 0; n < 16; ++n)
      acc[n] = wmma_bf(a, ld_frag(of1 + (size_t)(n * 16 + lm) * H, kt * 32), acc[n]);
  }
#pragma unroll
  for (int n = 0; n < 16; ++n) {
#pragma unroll
    for (int j = 0; j < 8; ++j) {
      const int rr = j + hi * 8, cc = n * 16 + lm;
      float v = acc[n][j] + of1b[cc];
      p1s[wid][rr * H + cc] = (__bf16)(v > 0.f ? v : 0.f);
      acc[n][j] = 0.f;
    }
  }
  __syncthreads();

  // of2
#pragma unroll 1
  for (int kt = 0; kt < 8; ++kt) {
    v16bf a = ld_frag(&p1s[wid][lm * H], kt * 32);
#pragma unroll
    for (int n = 0; n < 16; ++n)
      acc[n] = wmma_bf(a, ld_frag(of2 + (size_t)(n * 16 + lm) * H, kt * 32), acc[n]);
  }
#pragma unroll
  for (int n = 0; n < 16; ++n) {
#pragma unroll
    for (int j = 0; j < 8; ++j) {
      const int rr = j + hi * 8, cc = n * 16 + lm;
      float v = acc[n][j] + of2b[cc];
      v = v > 0.f ? v : 0.f;
      pred_out[(size_t)(row0 + rr) * H + cc] = v;
      p2s[wid][rr * H + cc] = (__bf16)v;
    }
  }
  __syncthreads();

  // mu (N = 64)
  v8f am[4];
#pragma unroll
  for (int n = 0; n < 4; ++n) am[n] = vzero;
#pragma unroll 1
  for (int kt = 0; kt < 8; ++kt) {
    v16bf a = ld_frag(&p2s[wid][lm * H], kt * 32);
#pragma unroll
    for (int n = 0; n < 4; ++n)
      am[n] = wmma_bf(a, ld_frag(muw + (size_t)(n * 16 + lm) * H, kt * 32), am[n]);
  }
#pragma unroll
  for (int n = 0; n < 4; ++n) {
#pragma unroll
    for (int j = 0; j < 8; ++j) {
      const int rr = j + hi * 8, cc = n * 16 + lm;
      mu_out[(size_t)(row0 + rr) * D + cc] = am[n][j] + mub[cc];
    }
  }
}

// ---------------------------------------------------------------------------
extern "C" void kernel_launch(void* const* d_in, const int* in_sizes, int n_in,
                              void* d_out, int out_size, void* d_ws, size_t ws_size,
                              hipStream_t stream)
{
  (void)in_sizes; (void)n_in; (void)out_size; (void)ws_size;
  const float* inputs = (const float*)d_in[0];
  const float* hidden = (const float*)d_in[1];
  const float* edges  = (const float*)d_in[2];
  const float* w1     = (const float*)d_in[3];
  const float* b1     = (const float*)d_in[4];
  const float* w2     = (const float*)d_in[5];
  const float* b2     = (const float*)d_in[6];
  const float* hr_w   = (const float*)d_in[7];
  const float* hi_w   = (const float*)d_in[8];
  const float* hh_w   = (const float*)d_in[9];
  const float* ir_w   = (const float*)d_in[10];
  const float* ir_b   = (const float*)d_in[11];
  const float* ii_w   = (const float*)d_in[12];
  const float* ii_b   = (const float*)d_in[13];
  const float* in_w   = (const float*)d_in[14];
  const float* in_b   = (const float*)d_in[15];
  const float* of1_w  = (const float*)d_in[16];
  const float* of1_b  = (const float*)d_in[17];
  const float* of2_w  = (const float*)d_in[18];
  const float* of2_b  = (const float*)d_in[19];
  const float* mu_w   = (const float*)d_in[20];
  const float* mu_b   = (const float*)d_in[21];

  float* out      = (float*)d_out;
  float* mu_out   = out;                           // [B,V,D]
  float* nh_out   = out + (size_t)ROWS_N * D;      // [B,V,H]
  float* pred_out = nh_out + (size_t)ROWS_N * H;   // [B,V,H]

  size_t off = 0;
  auto bump = [&](size_t bytes) -> void* {
    off = (off + 255) & ~(size_t)255;
    void* p = (char*)d_ws + off;
    off += bytes;
    return p;
  };
  float*  agg   = (float*) bump(sizeof(float) * (size_t)ROWS_N * H);
  __bf16* hidbf = (__bf16*)bump(2ull * ROWS_N * H);
  __bf16* inpbf = (__bf16*)bump(2ull * ROWS_N * D);
  __bf16* w1bf  = (__bf16*)bump(2ull * 4 * H * TWOH);
  __bf16* w2bf  = (__bf16*)bump(2ull * 4 * H * H);
  __bf16* hrbf  = (__bf16*)bump(2ull * H * H);
  __bf16* hibf  = (__bf16*)bump(2ull * H * H);
  __bf16* hhbf  = (__bf16*)bump(2ull * H * H);
  __bf16* irbf  = (__bf16*)bump(2ull * H * D);
  __bf16* iibf  = (__bf16*)bump(2ull * H * D);
  __bf16* inbf  = (__bf16*)bump(2ull * H * D);
  __bf16* of1bf = (__bf16*)bump(2ull * H * H);
  __bf16* of2bf = (__bf16*)bump(2ull * H * H);
  __bf16* mubf  = (__bf16*)bump(2ull * D * H);
  __bf16* nhbf  = (__bf16*)bump(2ull * ROWS_N * H);

  auto cvt = [&](const float* s, __bf16* d, int n) {
    k_f32_to_bf16<<<(n + 255) / 256, 256, 0, stream>>>(s, d, n);
  };
  k_zero<<<(ROWS_N * H + 255) / 256, 256, 0, stream>>>(agg, ROWS_N * H);
  cvt(hidden, hidbf, ROWS_N * H);
  cvt(inputs, inpbf, ROWS_N * D);
  cvt(w1,    w1bf, 4 * H * TWOH);
  cvt(w2,    w2bf, 4 * H * H);
  cvt(hr_w,  hrbf, H * H);
  cvt(hi_w,  hibf, H * H);
  cvt(hh_w,  hhbf, H * H);
  cvt(ir_w,  irbf, H * D);
  cvt(ii_w,  iibf, H * D);
  cvt(in_w,  inbf, H * D);
  cvt(of1_w, of1bf, H * H);
  cvt(of2_w, of2bf, H * H);
  cvt(mu_w,  mubf, D * H);

  // 2 waves (2 x 16-row tiles) per 64-thread block
  k_edge_msgs<<<ROWS_E / 32, 64, 0, stream>>>(hidbf, w1bf, b1, w2bf, b2, edges, agg);
  k_gru<<<ROWS_N / 32, 64, 0, stream>>>(agg, inpbf, hidden, hrbf, hibf, hhbf,
                                        irbf, ir_b, iibf, ii_b, inbf, in_b,
                                        nh_out, nhbf);
  k_outhead<<<ROWS_N / 32, 64, 0, stream>>>(nhbf, of1bf, of1_b, of2bf, of2_b,
                                            mubf, mu_b, pred_out, mu_out);
}
